// SNNBase_88364657148436
// MI455X (gfx1250) — matrix operand order repaired and checked by reading, not compiled, not run on hardware
//
#include <hip/hip_runtime.h>
#include <stdint.h>

// ---------------------------------------------------------------------------
// SNN forward (LIF, reset-by-subtract), CDNA5 / gfx1250, wave32.
//   Pre: x -> f16, W1/W2 -> f16 transposed [N][K] (one-time, lives in L2)
//   cur1 = x@W1 + b1                           (WMMA f16, fp32 accum, once)
//   10x: LIF1 -> s1@W2+b2 (WMMA, LIF2 + output-dot fused) -> tiny out-LIF
//   out = mean over steps of mo
// GEMM core: double-buffered LDS fed by global_load_async_to_lds_b128
// (ASYNCcnt), 128x128 block tile, 8 waves x 8 wmma per 32-K chunk,
// pipeline unrolled x2 so buffer addressing is static.
// ---------------------------------------------------------------------------

typedef __attribute__((ext_vector_type(16))) _Float16 v16h;
typedef __attribute__((ext_vector_type(8)))  _Float16 v8h;
typedef __attribute__((ext_vector_type(4)))  _Float16 v4h;
typedef __attribute__((ext_vector_type(8)))  float    v8f;
typedef __attribute__((ext_vector_type(4)))  float    v4f;

#define MT   128   // block tile rows (batch)
#define NT   128   // block tile cols
#define KC   32    // K chunk == WMMA K
#define LDA  32    // halves per As row (64B rows, 16B-aligned frags)
#define LDB  40    // halves per Bs row (80B rows: padded, 16B-aligned frags)

#define BETA  0.9f
#define THR   1.0f
#define NSTEP 10

// CDNA5 async global->LDS copy, 16B per lane (tracked by ASYNCcnt).
__device__ __forceinline__ void async_b128(uint32_t lds_off, const _Float16* g)
{
    asm volatile("global_load_async_to_lds_b128 %0, %1, off"
                 :: "v"(lds_off), "v"(g) : "memory");
}

// One K-chunk: load A/B fragments per CDNA5 16-bit WMMA layouts, 8 WMMAs.
__device__ __forceinline__ void mma_chunk(
    const _Float16* __restrict__ As, const _Float16* __restrict__ Bs,
    int waveM, int waveN, int r, int hi, v8f (&acc)[2][4])
{
    v16h af[2], bf[4];
    #pragma unroll
    for (int fm = 0; fm < 2; ++fm) {
        // A 16x32: lanes 0-15 K {0..7,16..23}; lanes 16-31 K {8..15,24..31}
        const _Float16* ap = As + (waveM * 32 + fm * 16 + r) * LDA + hi * 8;
        v8h lo8 = *(const v8h*)ap;
        v8h h8  = *(const v8h*)(ap + 16);
        #pragma unroll
        for (int i = 0; i < 8; ++i) { af[fm][i] = lo8[i]; af[fm][8 + i] = h8[i]; }
    }
    #pragma unroll
    for (int fn = 0; fn < 4; ++fn) {
        // B 32x16: lane holds one column; lanes 0-15 K {0..7,16..23}; lanes 16-31 K {8..15,24..31}
        const _Float16* bp = Bs + (waveN * 64 + fn * 16 + r) * LDB + hi * 8;
        v8h lo8 = *(const v8h*)bp;
        v8h h8  = *(const v8h*)(bp + 16);
        #pragma unroll
        for (int i = 0; i < 8; ++i) { bf[fn][i] = lo8[i]; bf[fn][8 + i] = h8[i]; }
    }
    #pragma unroll
    for (int fm = 0; fm < 2; ++fm)
        #pragma unroll
        for (int fn = 0; fn < 4; ++fn)
            acc[fm][fn] = __builtin_amdgcn_wmma_f32_16x16x32_f16(
                false, af[fm], false, bf[fn], (short)0, acc[fm][fn], false, false);
}

// ---------------------------------------------------------------------------
// GEMM + epilogue.  A[Bb][K] f16  x  Wt[N][K] f16 (pre-transposed) + bias.
// EPI==0: out_h = (f16)(C + bias)                        (cur1 path)
// EPI==1: LIF2 on m2, spike kept in registers, fused dot with Wo written as
//         deterministic per-(row, blockN, waveN) partials into dotp.
// ---------------------------------------------------------------------------
template<int EPI>
__global__ __launch_bounds__(256) void snn_gemm(
    const _Float16* __restrict__ A, const _Float16* __restrict__ Wt,
    const float* __restrict__ bias, _Float16* __restrict__ out_h,
    float* __restrict__ mem, const float* __restrict__ Wo,
    float* __restrict__ dotp, int Bb, int K, int N)
{
    __shared__ __align__(16) _Float16 smem[2 * MT * LDA + 2 * NT * LDB];

    const int t     = threadIdx.x;
    const int lane  = t & 31;
    const int wave  = t >> 5;       // 0..7
    const int r     = lane & 15;
    const int hi    = lane >> 4;    // half-wave select
    const int waveM = wave >> 1;    // 0..3  -> 32-row slice
    const int waveN = wave & 1;     // 0..1  -> 64-col slice

    const int b0 = blockIdx.y * MT;
    const int n0 = blockIdx.x * NT;

    // ---- per-thread staging addresses (2 A copies + 2 B copies, 16B each) --
    const int sRow = t >> 2;        // 0..63
    const int sC   = t & 3;         // 16B chunk within a 64B row
    const _Float16* aG0 = A  + (size_t)(b0 + sRow) * K + sC * 8;
    const _Float16* aG1 = aG0 + (size_t)64 * K;
    const _Float16* bG0 = Wt + (size_t)(n0 + sRow) * K + sC * 8;
    const _Float16* bG1 = bG0 + (size_t)64 * K;

    const uint32_t ldsBase = (uint32_t)(uintptr_t)smem;   // low 32 = LDS offset
    const uint32_t aSz = MT * LDA * 2;                    // 8 KB per buffer
    const uint32_t bSz = NT * LDB * 2;                    // 10 KB per buffer
    const uint32_t aOffA = ldsBase + (sRow * LDA + sC * 8) * 2;
    const uint32_t aOffB = aOffA + 64 * LDA * 2;
    const uint32_t bOffA = ldsBase + 2 * aSz + (sRow * LDB + sC * 8) * 2;
    const uint32_t bOffB = bOffA + 64 * LDB * 2;

    const _Float16* As0 = smem;
    const _Float16* As1 = smem + MT * LDA;
    const _Float16* Bs0 = smem + 2 * MT * LDA;
    const _Float16* Bs1 = smem + 2 * MT * LDA + NT * LDB;

    v8f acc[2][4] = {};

    const int nk = K / KC;          // 16 or 32 (always even)

    // prologue: chunk 0 -> buf0, chunk 1 -> buf1
    async_b128(aOffA, aG0); async_b128(aOffB, aG1);
    async_b128(bOffA, bG0); async_b128(bOffB, bG1);
    aG0 += KC; aG1 += KC; bG0 += KC; bG1 += KC;
    async_b128(aOffA + aSz, aG0); async_b128(aOffB + aSz, aG1);
    async_b128(bOffA + bSz, bG0); async_b128(bOffB + bSz, bG1);
    aG0 += KC; aG1 += KC; bG0 += KC; bG1 += KC;
    asm volatile("s_wait_asynccnt 0x4" ::: "memory");     // chunk 0 landed
    __syncthreads();

    for (int kk = 0; kk < nk; kk += 2) {
        const bool more = (kk + 2 < nk);

        // ---- consume chunk kk (buf0) ----
        mma_chunk(As0, Bs0, waveM, waveN, r, hi, acc);
        __syncthreads();                                   // done reading buf0
        if (more) {
            async_b128(aOffA, aG0); async_b128(aOffB, aG1);
            async_b128(bOffA, bG0); async_b128(bOffB, bG1);
            aG0 += KC; aG1 += KC; bG0 += KC; bG1 += KC;
            asm volatile("s_wait_asynccnt 0x4" ::: "memory"); // chunk kk+1 in
        } else {
            asm volatile("s_wait_asynccnt 0x0" ::: "memory");
        }
        __syncthreads();

        // ---- consume chunk kk+1 (buf1) ----
        mma_chunk(As1, Bs1, waveM, waveN, r, hi, acc);
        __syncthreads();                                   // done reading buf1
        if (more) {
            async_b128(aOffA + aSz, aG0); async_b128(aOffB + aSz, aG1);
            async_b128(bOffA + bSz, bG0); async_b128(bOffB + bSz, bG1);
            aG0 += KC; aG1 += KC; bG0 += KC; bG1 += KC;
            asm volatile("s_wait_asynccnt 0x4" ::: "memory"); // chunk kk+2 in
        } else {
            asm volatile("s_wait_asynccnt 0x0" ::: "memory");
        }
        __syncthreads();
    }

    // ---- epilogue: C/D layout: VGPR i -> M = i + hi*8, N = lane&15 ----
    if (EPI == 0) {
        #pragma unroll
        for (int fm = 0; fm < 2; ++fm) {
            #pragma unroll
            for (int fn = 0; fn < 4; ++fn) {
                int col = n0 + waveN * 64 + fn * 16 + r;
                float bv = bias[col];
                #pragma unroll
                for (int i = 0; i < 8; ++i) {
                    int row = b0 + waveM * 32 + fm * 16 + hi * 8 + i;
                    out_h[(size_t)row * N + col] = (_Float16)(acc[fm][fn][i] + bv);
                }
            }
        }
    } else {
        #pragma unroll
        for (int fm = 0; fm < 2; ++fm) {
            float p[8] = {0.f, 0.f, 0.f, 0.f, 0.f, 0.f, 0.f, 0.f};
            #pragma unroll
            for (int fn = 0; fn < 4; ++fn) {
                int col = n0 + waveN * 64 + fn * 16 + r;
                float bv = bias[col];
                float wo = Wo[col];
                #pragma unroll
                for (int i = 0; i < 8; ++i) {
                    int row = b0 + waveM * 32 + fm * 16 + hi * 8 + i;
                    size_t idx = (size_t)row * N + col;
                    float c     = acc[fm][fn][i] + bv;
                    float mold  = mem[idx];
                    float reset = (mold > THR) ? THR : 0.0f;
                    float mnew  = BETA * mold + c - reset;
                    mem[idx]    = mnew;
                    float s     = (mnew - THR > 0.0f) ? 1.0f : 0.0f;
                    p[i] += s * wo;                 // fused output-layer dot
                }
            }
            // reduce each row partial across the 16 lanes of this half-wave
            #pragma unroll
            for (int i = 0; i < 8; ++i) {
                float v = p[i];
                #pragma unroll
                for (int off = 1; off < 16; off <<= 1)
                    v += __shfl_xor(v, off, 32);
                if (r == 0) {
                    int row = b0 + waveM * 32 + fm * 16 + hi * 8 + i;
                    dotp[(size_t)row * 16 + blockIdx.x * 2 + waveN] = v;
                }
            }
        }
    }
}

// ---------------------------------------------------------------------------
// One-time conversions: x -> f16 ;  W[K][N] f32 -> Wt[N][K] f16 (transposed)
// ---------------------------------------------------------------------------
__global__ __launch_bounds__(256) void conv_h(const float* __restrict__ src,
                                              _Float16* __restrict__ dst, size_t n)
{
    size_t i = ((size_t)blockIdx.x * blockDim.x + threadIdx.x) * 4;
    if (i >= n) return;
    v4f v = *(const v4f*)(src + i);
    v4h h;
    #pragma unroll
    for (int j = 0; j < 4; ++j) h[j] = (_Float16)v[j];
    *(v4h*)(dst + i) = h;
}

__global__ void transp_h(const float* __restrict__ W, _Float16* __restrict__ Wt,
                         int K, int N)
{
    __shared__ _Float16 tile[32][33];
    int k0 = blockIdx.y * 32, n0 = blockIdx.x * 32;
    int tx = threadIdx.x, ty = threadIdx.y;        // (32, 8)
    #pragma unroll
    for (int s = 0; s < 32; s += 8)
        tile[ty + s][tx] = (_Float16)W[(size_t)(k0 + ty + s) * N + n0 + tx];
    __syncthreads();
    #pragma unroll
    for (int s = 0; s < 32; s += 8)
        Wt[(size_t)(n0 + ty + s) * K + k0 + tx] = tile[tx][ty + s];
}

// ---------------------------------------------------------------------------
// LIF layer 1 (elementwise, 4-wide)
// ---------------------------------------------------------------------------
__global__ __launch_bounds__(256) void snn_lif1(
    const _Float16* __restrict__ cur1, float* __restrict__ m1,
    _Float16* __restrict__ s1, size_t total)
{
    size_t i = ((size_t)blockIdx.x * blockDim.x + threadIdx.x) * 4;
    if (i >= total) return;
    v4h c = *(const v4h*)(cur1 + i);
    v4f m = *(const v4f*)(m1 + i);
    v4h s;
    #pragma unroll
    for (int j = 0; j < 4; ++j) {
        float mold  = m[j];
        float reset = (mold > THR) ? THR : 0.0f;
        float mnew  = BETA * mold + (float)c[j] - reset;
        m[j] = mnew;
        s[j] = (mnew - THR > 0.0f) ? (_Float16)1.0f : (_Float16)0.0f;
    }
    *(v4f*)(m1 + i) = m;
    *(v4h*)(s1 + i) = s;
}

// ---------------------------------------------------------------------------
// Output-layer LIF: sum the 16 deterministic dot partials per row, update mo.
// ---------------------------------------------------------------------------
__global__ __launch_bounds__(256) void snn_outlif(
    const float* __restrict__ dotp, const float* __restrict__ bo,
    float* __restrict__ mo, float* __restrict__ acc, int Bb, int parts)
{
    int b = blockIdx.x * 256 + threadIdx.x;
    if (b >= Bb) return;
    const float* d = dotp + (size_t)b * 16;
    float s = 0.0f;
    for (int j = 0; j < parts; ++j) s += d[j];
    float cur   = s + bo[0];
    float mold  = mo[b];
    float reset = (mold > THR) ? THR : 0.0f;
    float mnew  = BETA * mold + cur - reset;
    mo[b]  = mnew;
    acc[b] += mnew;
}

__global__ void snn_zero(float* __restrict__ p, size_t n)
{
    size_t i = (size_t)blockIdx.x * blockDim.x + threadIdx.x;
    size_t stride = (size_t)gridDim.x * blockDim.x;
    for (; i < n; i += stride) p[i] = 0.0f;
}

__global__ void snn_fin(const float* __restrict__ acc, float* __restrict__ out, int Bb)
{
    int i = blockIdx.x * blockDim.x + threadIdx.x;
    if (i < Bb) out[i] = acc[i] * (1.0f / (float)NSTEP);
}

// ---------------------------------------------------------------------------
extern "C" void kernel_launch(void* const* d_in, const int* in_sizes, int n_in,
                              void* d_out, int out_size, void* d_ws, size_t ws_size,
                              hipStream_t stream)
{
    const float* x  = (const float*)d_in[0];
    const float* W1 = (const float*)d_in[1];
    const float* b1 = (const float*)d_in[2];
    const float* W2 = (const float*)d_in[3];
    const float* b2 = (const float*)d_in[4];
    const float* Wo = (const float*)d_in[5];
    const float* bo = (const float*)d_in[6];
    float* out = (float*)d_out;

    const int H  = in_sizes[2];            // 1024
    const int IN = in_sizes[1] / H;        // 512
    const int Bb = in_sizes[0] / IN;       // 65536

    // ws: [m1][m2][mo][acc][dotp] f32 | [cur1][s1][xh] f16 | [W1t][W2t] f16
    char* ws = (char*)d_ws;
    float* m1   = (float*)ws;       ws += (size_t)Bb * H * sizeof(float);
    float* m2   = (float*)ws;       ws += (size_t)Bb * H * sizeof(float);
    float* mo   = (float*)ws;       ws += (size_t)Bb * sizeof(float);
    float* accb = (float*)ws;       ws += (size_t)Bb * sizeof(float);
    float* dotp = (float*)ws;       ws += (size_t)Bb * 16 * sizeof(float);
    _Float16* cur1 = (_Float16*)ws; ws += (size_t)Bb * H * sizeof(_Float16);
    _Float16* s1   = (_Float16*)ws; ws += (size_t)Bb * H * sizeof(_Float16);
    _Float16* xh   = (_Float16*)ws; ws += (size_t)Bb * IN * sizeof(_Float16);
    _Float16* W1t  = (_Float16*)ws; ws += (size_t)H * IN * sizeof(_Float16);
    _Float16* W2t  = (_Float16*)ws; ws += (size_t)H * H * sizeof(_Float16);

    // zero persistent state (m1, m2, mo, acc, dotp are contiguous)
    size_t zn = (size_t)2 * Bb * H + (size_t)18 * Bb;
    snn_zero<<<2048, 256, 0, stream>>>(m1, zn);

    // one-time f16 conversions / transposes (W1t, W2t stay hot in 192MB L2)
    size_t xn = (size_t)Bb * IN;
    conv_h<<<(int)(xn / 4 / 256), 256, 0, stream>>>(x, xh, xn);
    transp_h<<<dim3(H / 32, IN / 32), dim3(32, 8), 0, stream>>>(W1, W1t, IN, H);
    transp_h<<<dim3(H / 32, H / 32),  dim3(32, 8), 0, stream>>>(W2, W2t, H, H);

    dim3 gg(H / NT, Bb / MT);   // (8, 512)
    const int parts = (H / NT) * 2;  // 16 dot partials per row

    // cur1 = x @ W1 + b1  (input current, identical every step)
    snn_gemm<0><<<gg, 256, 0, stream>>>(xh, W1t, b1, cur1, nullptr, nullptr,
                                        nullptr, Bb, IN, H);

    size_t total = (size_t)Bb * H;
    int l1blocks = (int)(total / 4 / 256);
    for (int t = 0; t < NSTEP; ++t) {
        snn_lif1<<<l1blocks, 256, 0, stream>>>(cur1, m1, s1, total);
        snn_gemm<1><<<gg, 256, 0, stream>>>(s1, W2t, b2, nullptr, m2, Wo,
                                            dotp, Bb, H, H);
        snn_outlif<<<(Bb + 255) / 256, 256, 0, stream>>>(dotp, bo, mo, accb,
                                                         Bb, parts);
    }
    snn_fin<<<(Bb + 255) / 256, 256, 0, stream>>>(accb, out, Bb);
}
